// e3nnNetwork_41790031790112
// MI455X (gfx1250) — compile-verified, matrix operand order checked by reading, not run on hardware
//
#include <hip/hip_runtime.h>
#include <hip/hip_bf16.h>
#include <cmath>

typedef _Float16 v16h __attribute__((ext_vector_type(16)));
typedef _Float16 v8h  __attribute__((ext_vector_type(8)));
typedef float    v8f  __attribute__((ext_vector_type(8)));

#define NGc   32
#define GPGc  128
#define NHc   4096      // nodes per half
#define NNc   8192      // total nodes
#define EEc   262144    // edges (N*K)

// constants
#define CSc     0.3826834323650898f   // sin(pi/8)
#define CXc     0.9238795325112867f   // cos(pi/8)
#define S3c     1.7320508075688772f
#define S5c     2.2360679774997896f
#define S7c     2.6457513110645907f
#define S15c    3.8729833462074170f
#define C9c     1.0801234497346435f   // sqrt(42)/6
#define C11c    1.6201851746019650f   // sqrt(168)/8
#define SQ10c   3.1622776601683795f
#define ISQ10c  0.31622776601683794f
#define ISQ128c 0.08838834764831845f
#define ISQ32c  0.17677669529663687f
#define ISQ3c   0.5773502691896258f
#define ISQ2c   0.7071067811865476f
#define ISQ5c   0.4472135954999579f
#define ISQ7c   0.3779644730092272f

// fragment-array layout (halfs) inside the staged weight region
#define FO_W1A  0        // 8 nt  * 32 lanes * 16        = 4096
#define FO_W1B  4096     // 7 nt2 * 4 ks * 32 * 16       = 14336
#define FO_W2A  18432    // 8 nt  * 32 * 16              = 4096
#define FO_W2B  22528    // 4 ks  * 32 * 16              = 2048
#define FRAG_TOT 24576

__device__ __forceinline__ float siluf(float v){
  return v * __builtin_amdgcn_rcpf(1.f + __expf(-v));
}
__device__ __forceinline__ float sigmf(float v){
  return __builtin_amdgcn_rcpf(1.f + __expf(-v));
}
// K index held by vector element i for half (lane>>4) in the 16-bit 16x32 A/B layout
__device__ __forceinline__ int kOf(int i, int half){
  return (i < 8) ? (half*8 + i) : (16 + half*8 + (i - 8));
}

// -------- K0: pre-swizzle weights into WMMA B-fragment-major f16 layout --------
__global__ void k_stage_frags(const float* __restrict__ fc1_w1, const float* __restrict__ fc1_w2,
                              const float* __restrict__ fc2_w1, const float* __restrict__ fc2_w2,
                              _Float16* __restrict__ frag){
  int idx = blockIdx.x*blockDim.x + threadIdx.x;
  if (idx >= FRAG_TOT) return;
  float val;
  if (idx < FO_W1B){                       // fc1_w1 (10x128), K padded to 32
    int nt = idx >> 9, lane = (idx >> 4) & 31, i = idx & 15;
    int half = lane >> 4, c = lane & 15, k = kOf(i, half);
    val = (k < 10) ? fc1_w1[k*128 + nt*16 + c] : 0.f;
  } else if (idx < FO_W2A){                // fc1_w2 (128x100), N padded to 112
    int l = idx - FO_W1B;
    int nt2 = l >> 11, ks = (l >> 9) & 3, lane = (l >> 4) & 31, i = l & 15;
    int half = lane >> 4, c = lane & 15, k = ks*32 + kOf(i, half), cg = nt2*16 + c;
    val = (cg < 100) ? fc1_w2[k*100 + cg] : 0.f;
  } else if (idx < FO_W2B){                // fc2_w1 (10x128)
    int l = idx - FO_W2A;
    int nt = l >> 9, lane = (l >> 4) & 31, i = l & 15;
    int half = lane >> 4, c = lane & 15, k = kOf(i, half);
    val = (k < 10) ? fc2_w1[k*128 + nt*16 + c] : 0.f;
  } else {                                 // fc2_w2 (128x16)
    int l = idx - FO_W2B;
    int ks = l >> 9, lane = (l >> 4) & 31, i = l & 15;
    int half = lane >> 4, c = lane & 15, k = ks*32 + kOf(i, half);
    val = fc2_w2[k*16 + c];
  }
  frag[idx] = (_Float16)val;
}

// ---------------- K1: node linears: h = x@lin1a/5, sc1 = x@sc1_w/5 ----------------
__global__ void k_node_lin(const float* __restrict__ wt_x, const float* __restrict__ mt_x,
                           const float* __restrict__ sc1_w, const float* __restrict__ lin1a,
                           float* __restrict__ hbuf, float* __restrict__ sc1buf){
  int n = blockIdx.x*blockDim.x + threadIdx.x;
  if (n >= NNc) return;
  const float* xp = (n < NHc) ? (wt_x + (size_t)n*25) : (mt_x + (size_t)(n-NHc)*25);
  float xv[25];
  #pragma unroll
  for (int u=0;u<25;++u) xv[u] = xp[u];
  for (int v=0;v<16;++v){
    float a=0.f;
    #pragma unroll
    for (int u=0;u<25;++u) a += xv[u]*sc1_w[u*16+v];
    sc1buf[(size_t)n*16+v] = a*0.2f;
  }
  for (int v=0;v<25;++v){
    float a=0.f;
    #pragma unroll
    for (int u=0;u<25;++u) a += xv[u]*lin1a[u*25+v];
    hbuf[(size_t)n*25+v] = a*0.2f;
  }
}

// ---------------- K2: per-edge geometry: sh[16] (f32), emb (f16, padded to 32) ----------------
__global__ void k_edge_geom(const float* __restrict__ wt_pos, const float* __restrict__ mt_pos,
                            const int* __restrict__ esrc, const int* __restrict__ edst,
                            float* __restrict__ shb, _Float16* __restrict__ embF){
  int e = blockIdx.x*blockDim.x + threadIdx.x;
  if (e >= EEc) return;
  int s = esrc[e], d = edst[e];
  const float* ps = (s < NHc) ? (wt_pos + 3*(size_t)s) : (mt_pos + 3*(size_t)(s-NHc));
  const float* pd = (d < NHc) ? (wt_pos + 3*(size_t)d) : (mt_pos + 3*(size_t)(d-NHc));
  float vx=ps[0]-pd[0], vy=ps[1]-pd[1], vz=ps[2]-pd[2];
  float r = sqrtf(vx*vx+vy*vy+vz*vz);
  float ir = __builtin_amdgcn_rcpf(fmaxf(r, 1e-9f));
  float x=vx*ir, y=vy*ir, z=vz*ir;
  float x2=x*x, y2=y*y, z2=z*z;
  float sh4 = S15c*x*z;             // sh2_0
  float sh8 = 0.5f*S15c*(z2-x2);    // sh2_4
  float sh[16];
  sh[0]=1.f; sh[1]=S3c*x; sh[2]=S3c*y; sh[3]=S3c*z;
  sh[4]=sh4; sh[5]=S15c*x*y; sh[6]=S5c*(y2-0.5f*(x2+z2)); sh[7]=S15c*y*z; sh[8]=sh8;
  sh[9]  = C9c*(sh4*z + sh8*x);
  sh[10] = S7c*sh4*y;
  sh[11] = C11c*(4.f*y2-x2-z2)*x;
  sh[12] = 0.5f*S7c*y*(2.f*y2-3.f*x2-3.f*z2);
  sh[13] = C11c*z*(4.f*y2-x2-z2);
  sh[14] = S7c*sh8*y;
  sh[15] = C9c*(sh8*z - sh4*x);
  float* shp = shb + (size_t)e*16;
  #pragma unroll
  for (int i=0;i<16;++i) shp[i] = sh[i];
  // radial gaussian basis, f16, zero-padded to 32 (A-fragment friendly)
  const float step = 20.0f/9.0f;
  const float istep = 9.0f/20.0f;
  v8h c0, c1, c2, c3;
  #pragma unroll
  for (int j=0;j<8;++j){
    float t = (r - (float)j*step)*istep;
    c0[j] = (_Float16)(__expf(-t*t)*SQ10c);
  }
  #pragma unroll
  for (int j=0;j<8;++j){
    int jj = 8+j;
    float t = (r - (float)jj*step)*istep;
    c1[j] = (jj < 10) ? (_Float16)(__expf(-t*t)*SQ10c) : (_Float16)0.f;
    c2[j] = (_Float16)0.f;
    c3[j] = (_Float16)0.f;
  }
  _Float16* ep = embF + (size_t)e*32;
  *(v8h*)(ep)      = c0;
  *(v8h*)(ep + 8)  = c1;
  *(v8h*)(ep + 16) = c2;
  *(v8h*)(ep + 24) = c3;
}

// -------- K3: WMMA edge MLPs, fused message-1 scatter. 4 waves/block, 1 tile(16 edges)/wave --------
__global__ __launch_bounds__(128) void k_edge_mlp_wmma(
    const _Float16* __restrict__ embF, const float* __restrict__ shb,
    const int* __restrict__ esrc, const int* __restrict__ edst,
    const float* __restrict__ hbuf,
    const _Float16* __restrict__ frag,
    float* __restrict__ mall, float* __restrict__ w2out){
  __shared__ __attribute__((aligned(32))) _Float16 s_t[4][16*128];

  const int lane = threadIdx.x & 31;
  const int wid  = threadIdx.x >> 5;
  const int half = lane >> 4;
  const int col  = lane & 15;        // M-row for A frags / N-column for B,C,D frags
  const int tile = blockIdx.x*4 + wid;
  const int e0 = tile*16;
  _Float16* tb = &s_t[wid][0];

  const _Float16* fW1a = frag + FO_W1A;
  const _Float16* fW1b = frag + FO_W1B;
  const _Float16* fW2a = frag + FO_W2A;
  const _Float16* fW2b = frag + FO_W2B;

  // A fragment of radial embedding: 16 edges x 32 K (two contiguous 16B chunks per lane)
  const _Float16* ep = embF + (size_t)(e0+col)*32;
  v8h elo = *(const v8h*)(ep + half*8);
  v8h ehi = *(const v8h*)(ep + 16 + half*8);
  v16h aE;
  #pragma unroll
  for (int i=0;i<8;++i){ aE[i]=elo[i]; aE[8+i]=ehi[i]; }

  int eI[8], sI[8], dI[8];
  #pragma unroll
  for (int v=0; v<8; ++v){
    int e = e0 + v + 8*half;
    eI[v]=e; sI[v]=esrc[e]; dI[v]=edst[e];
  }

  // ---- stage 1a: t1 = silu(emb @ fc1_w1 / sqrt(10)) : 8 WMMAs ----
  #pragma unroll
  for (int nt=0; nt<8; ++nt){
    v16h b = *(const v16h*)(fW1a + ((size_t)(nt*32 + lane))*16);
    v8f c = {};
    c = __builtin_amdgcn_wmma_f32_16x16x32_f16(false, aE, false, b, (short)0, c, false, false);
    #pragma unroll
    for (int v=0;v<8;++v)
      tb[(v + 8*half)*128 + nt*16 + col] = (_Float16)siluf(c[v]*ISQ10c);
  }
  asm volatile("s_wait_dscnt 0" ::: "memory");

  // A fragments of t1 for the K=128 contraction (4 K-slabs, vector LDS loads)
  v16h aT[4];
  #pragma unroll
  for (int ks=0; ks<4; ++ks){
    v8h lo = *(const v8h*)(tb + col*128 + ks*32 + half*8);
    v8h hi = *(const v8h*)(tb + col*128 + ks*32 + 16 + half*8);
    #pragma unroll
    for (int i=0;i<8;++i){ aT[ks][i]=lo[i]; aT[ks][8+i]=hi[i]; }
  }

  // ---- stage 1b: w1 = t1 @ fc1_w2 / sqrt(128): 28 WMMAs, fused scatter into m0..m3 ----
  #pragma unroll
  for (int nt2=0; nt2<7; ++nt2){
    const int colg = nt2*16 + col;   // 0..111, valid < 100
    v8f acc = {};
    #pragma unroll
    for (int ks=0; ks<4; ++ks){
      v16h bb = *(const v16h*)(fW1b + ((size_t)((nt2*4 + ks)*32 + lane))*16);
      acc = __builtin_amdgcn_wmma_f32_16x16x32_f16(false, aT[ks], false, bb, (short)0, acc, false, false);
    }
    if (colg < 100){
      const int blk = colg/25, u = colg%25;       // w1[:, blk, u]
      const int c0 = (blk==0)?0 : (blk==1)?1 : (blk==2)?4 : 9;
      const int ns = (blk==0)?1 : (blk==1)?3 : (blk==2)?5 : 7;
      #pragma unroll
      for (int v=0;v<8;++v){
        float w1v = acc[v]*ISQ128c;
        float p = hbuf[(size_t)sI[v]*25 + u] * w1v;
        float* mrow = mall + (size_t)dI[v]*400 + u*16;
        if (blk==0){
          atomicAdd(&mrow[0], p);
        } else {
          const float* sp = shb + (size_t)eI[v]*16;
          for (int s=0;s<ns;++s) atomicAdd(&mrow[c0+s], p*sp[c0+s]);
        }
      }
    }
  }
  asm volatile("s_wait_dscnt 0" ::: "memory");

  // ---- stage 2a: t2 = silu(emb @ fc2_w1 / sqrt(10)) : 8 WMMAs ----
  #pragma unroll
  for (int nt=0; nt<8; ++nt){
    v16h b = *(const v16h*)(fW2a + ((size_t)(nt*32 + lane))*16);
    v8f c = {};
    c = __builtin_amdgcn_wmma_f32_16x16x32_f16(false, aE, false, b, (short)0, c, false, false);
    #pragma unroll
    for (int v=0;v<8;++v)
      tb[(v + 8*half)*128 + nt*16 + col] = (_Float16)siluf(c[v]*ISQ10c);
  }
  asm volatile("s_wait_dscnt 0" ::: "memory");

  // ---- stage 2b: w2 = t2 @ fc2_w2 / sqrt(128): 4 WMMAs ----
  #pragma unroll
  for (int ks=0; ks<4; ++ks){
    v8h lo = *(const v8h*)(tb + col*128 + ks*32 + half*8);
    v8h hi = *(const v8h*)(tb + col*128 + ks*32 + 16 + half*8);
    #pragma unroll
    for (int i=0;i<8;++i){ aT[ks][i]=lo[i]; aT[ks][8+i]=hi[i]; }
  }
  v8f ac2 = {};
  #pragma unroll
  for (int ks=0; ks<4; ++ks){
    v16h bb = *(const v16h*)(fW2b + ((size_t)(ks*32 + lane))*16);
    ac2 = __builtin_amdgcn_wmma_f32_16x16x32_f16(false, aT[ks], false, bb, (short)0, ac2, false, false);
  }
  #pragma unroll
  for (int v=0;v<8;++v)
    w2out[(size_t)eI[v]*16 + col] = ac2[v]*ISQ128c;
}

// ---------------- K5: node update ----------------
__global__ void k_node_update(const float* __restrict__ mall, const float* __restrict__ sc1buf,
    const float* __restrict__ l2a0, const float* __restrict__ l2a1,
    const float* __restrict__ l2a2, const float* __restrict__ l2a3,
    const float* __restrict__ l1b0, const float* __restrict__ l1b1,
    const float* __restrict__ l1b2, const float* __restrict__ l1b3,
    float* __restrict__ nodeb){
  int n = blockIdx.x*blockDim.x + threadIdx.x;
  if (n >= NNc) return;
  float a0[16]={0}, a1[9]={0}, a2[10]={0}, a3[7]={0};
  const float* m = mall + (size_t)n*400;
  for (int u=0;u<25;++u){
    float row[16];
    #pragma unroll
    for (int c=0;c<16;++c) row[c] = m[u*16+c]*ISQ32c;   // seg /sqrt(NNEI)
    float m0 = row[0];
    #pragma unroll
    for (int v=0;v<16;++v) a0[v] += m0*l2a0[u*16+v];
    #pragma unroll
    for (int v=0;v<3;++v){ float w=l2a1[u*3+v];
      #pragma unroll
      for (int mm=0;mm<3;++mm) a1[v*3+mm] += row[1+mm]*w; }
    #pragma unroll
    for (int v=0;v<2;++v){ float w=l2a2[u*2+v];
      #pragma unroll
      for (int mm=0;mm<5;++mm) a2[v*5+mm] += row[4+mm]*w; }
    { float w=l2a3[u];
      #pragma unroll
      for (int mm=0;mm<7;++mm) a3[mm] += row[9+mm]*w; }
  }
  float scal[10], gates[6];
  #pragma unroll
  for (int j=0;j<16;++j){
    float s0 = CSc*sc1buf[(size_t)n*16+j] + CXc*(a0[j]*0.2f);
    if (j<10) scal[j]=siluf(s0); else gates[j-10]=sigmf(s0);
  }
  float g1[9], g2[10], g3[7];
  #pragma unroll
  for (int v=0;v<3;++v) for (int mm=0;mm<3;++mm) g1[v*3+mm] = a1[v*3+mm]*0.2f*gates[v];
  #pragma unroll
  for (int v=0;v<2;++v) for (int mm=0;mm<5;++mm) g2[v*5+mm] = a2[v*5+mm]*0.2f*gates[3+v];
  #pragma unroll
  for (int mm=0;mm<7;++mm) g3[mm] = a3[mm]*0.2f*gates[5];

  float* nb = nodeb + (size_t)n*46;
  #pragma unroll
  for (int j=0;j<10;++j) nb[j] = scal[j];
  for (int v=0;v<10;++v){
    float a=0.f;
    #pragma unroll
    for (int u=0;u<10;++u) a += scal[u]*l1b0[u*10+v];
    nb[10+v] = a*ISQ10c;
  }
  #pragma unroll
  for (int v=0;v<3;++v) for (int mm=0;mm<3;++mm){
    float a=0.f;
    #pragma unroll
    for (int u=0;u<3;++u) a += g1[u*3+mm]*l1b1[u*3+v];
    nb[20+v*3+mm] = a*ISQ3c;
  }
  #pragma unroll
  for (int v=0;v<2;++v) for (int mm=0;mm<5;++mm){
    float a=0.f;
    #pragma unroll
    for (int u=0;u<2;++u) a += g2[u*5+mm]*l1b2[u*2+v];
    nb[29+v*5+mm] = a*ISQ2c;
  }
  float b3 = l1b3[0];
  #pragma unroll
  for (int mm=0;mm<7;++mm) nb[39+mm] = g3[mm]*b3;
}

// ---------------- K6: second message pass ----------------
__global__ void k_edge_msg2(const int* __restrict__ esrc, const int* __restrict__ edst,
                            const float* __restrict__ shb, const float* __restrict__ w2b,
                            const float* __restrict__ nodeb, float* __restrict__ mid){
  int e = blockIdx.x*blockDim.x + threadIdx.x;
  if (e >= EEc) return;
  int s = esrc[e], d = edst[e];
  const float* sh = shb + (size_t)e*16;
  const float* w2 = w2b + (size_t)e*16;
  const float* nb = nodeb + (size_t)s*46;
  float out[16];
  #pragma unroll
  for (int u=0;u<10;++u) out[u] = nb[10+u]*w2[u];
  #pragma unroll
  for (int u=0;u<3;++u){
    float a=0.f;
    #pragma unroll
    for (int mm=0;mm<3;++mm) a += nb[20+u*3+mm]*sh[1+mm];
    out[10+u] = a*ISQ3c*w2[10+u];
  }
  #pragma unroll
  for (int u=0;u<2;++u){
    float a=0.f;
    #pragma unroll
    for (int mm=0;mm<5;++mm) a += nb[29+u*5+mm]*sh[4+mm];
    out[13+u] = a*ISQ5c*w2[13+u];
  }
  { float a=0.f;
    #pragma unroll
    for (int mm=0;mm<7;++mm) a += nb[39+mm]*sh[9+mm];
    out[15] = a*ISQ7c*w2[15]; }
  float* mp = mid + (size_t)d*16;
  #pragma unroll
  for (int c=0;c<16;++c) atomicAdd(&mp[c], out[c]);
}

// ---------------- K7: y = CS*sc2 + CX*(mid@lin2b)/4, graph-sum ----------------
__global__ void k_reduce(const float* __restrict__ nodeb, const float* __restrict__ mid,
                         const float* __restrict__ sc2_w, const float* __restrict__ lin2b,
                         float* __restrict__ ssum){
  int idx = blockIdx.x*blockDim.x + threadIdx.x;
  if (idx >= NNc*256) return;
  int n = idx >> 8, j = idx & 255;
  const float* nb = nodeb + (size_t)n*46;
  float sc2 = 0.f;
  #pragma unroll
  for (int u=0;u<10;++u) sc2 += nb[u]*sc2_w[u*256+j];
  sc2 *= ISQ10c;
  float mv = 0.f;
  #pragma unroll
  for (int u=0;u<16;++u) mv += mid[(size_t)n*16+u]*lin2b[u*256+j];
  mv *= ISQ32c;                      // seg /sqrt(NNEI)
  float y = CSc*sc2 + CXc*mv*0.25f;
  int g = (n & (NHc-1)) >> 7;        // batch id, GPG=128
  atomicAdd(&ssum[g*256+j], y);
}

// ---------------- K8: head ----------------
__global__ void k_head(const float* __restrict__ ssum, const float* __restrict__ head_w,
                       const float* __restrict__ head_b, float* __restrict__ out){
  int t = threadIdx.x;
  if (t >= 64) return;
  int o = t >> 5, g = t & 31;
  float a = 0.f;
  for (int j=0;j<256;++j) a += (ssum[g*256+j]*(1.f/256.f))*head_w[j*2+o];
  out[o*32+g] = a + head_b[o];
}

extern "C" void kernel_launch(void* const* d_in, const int* in_sizes, int n_in,
                              void* d_out, int out_size, void* d_ws, size_t ws_size,
                              hipStream_t stream){
  (void)in_sizes; (void)n_in; (void)out_size; (void)ws_size;
  const float* wt_pos=(const float*)d_in[0];
  const float* mt_pos=(const float*)d_in[1];
  const float* wt_x  =(const float*)d_in[2];
  const float* mt_x  =(const float*)d_in[3];
  const int*   esrc  =(const int*)  d_in[6];
  const int*   edst  =(const int*)  d_in[7];
  const float* sc1_w =(const float*)d_in[8];
  const float* lin1a =(const float*)d_in[9];
  const float* fc1_w1=(const float*)d_in[10];
  const float* fc1_w2=(const float*)d_in[11];
  const float* l2a0  =(const float*)d_in[12];
  const float* l2a1  =(const float*)d_in[13];
  const float* l2a2  =(const float*)d_in[14];
  const float* l2a3  =(const float*)d_in[15];
  const float* sc2_w =(const float*)d_in[16];
  const float* l1b0  =(const float*)d_in[17];
  const float* l1b1  =(const float*)d_in[18];
  const float* l1b2  =(const float*)d_in[19];
  const float* l1b3  =(const float*)d_in[20];
  const float* fc2_w1=(const float*)d_in[21];
  const float* fc2_w2=(const float*)d_in[22];
  const float* lin2b =(const float*)d_in[23];
  const float* head_w=(const float*)d_in[24];
  const float* head_b=(const float*)d_in[25];
  float* ws  = (float*)d_ws;
  float* out = (float*)d_out;

  // workspace layout (floats; every region size is a multiple of 8 floats = 32B aligned)
  const size_t o_sh   = 0;                               // E*16 f32
  const size_t o_emb  = o_sh   + (size_t)EEc*16;         // E*32 f16 (= E*16 f32 slots)
  const size_t o_h    = o_emb  + (size_t)EEc*16;         // N*25 f32
  const size_t o_sc1  = o_h    + (size_t)NNc*25;         // N*16 f32
  const size_t o_w2   = o_sc1  + (size_t)NNc*16;         // E*16 f32
  const size_t o_frag = o_w2   + (size_t)EEc*16;         // 24576 f16 (= 12288 f32 slots)
  const size_t o_m    = o_frag + 12288;                  // N*400 f32 (zeroed)
  const size_t o_mid  = o_m    + (size_t)NNc*400;        // N*16  f32 (zeroed)
  const size_t o_ss   = o_mid  + (size_t)NNc*16;         // NG*256 f32 (zeroed)
  const size_t o_nb   = o_ss   + (size_t)NGc*256;        // N*46 f32

  // zero accumulators (m_all, mid, ssum are contiguous)
  hipMemsetAsync(ws + o_m, 0, ((size_t)NNc*416 + (size_t)NGc*256)*sizeof(float), stream);

  k_stage_frags  <<<(FRAG_TOT+255)/256, 256, 0, stream>>>(fc1_w1, fc1_w2, fc2_w1, fc2_w2,
                                                          (_Float16*)(ws + o_frag));
  k_node_lin     <<<NNc/128, 128, 0, stream>>>(wt_x, mt_x, sc1_w, lin1a, ws+o_h, ws+o_sc1);
  k_edge_geom    <<<EEc/256, 256, 0, stream>>>(wt_pos, mt_pos, esrc, edst,
                                               ws+o_sh, (_Float16*)(ws+o_emb));
  k_edge_mlp_wmma<<<EEc/64,  128, 0, stream>>>((const _Float16*)(ws+o_emb), ws+o_sh,
                                               esrc, edst, ws+o_h,
                                               (const _Float16*)(ws+o_frag),
                                               ws+o_m, ws+o_w2);
  k_node_update  <<<NNc/128, 128, 0, stream>>>(ws+o_m, ws+o_sc1, l2a0, l2a1, l2a2, l2a3,
                                               l1b0, l1b1, l1b2, l1b3, ws+o_nb);
  k_edge_msg2    <<<EEc/256, 256, 0, stream>>>(esrc, edst, ws+o_sh, ws+o_w2, ws+o_nb, ws+o_mid);
  k_reduce       <<<(NNc*256)/256, 256, 0, stream>>>(ws+o_nb, ws+o_mid, sc2_w, lin2b, ws+o_ss);
  k_head         <<<1, 64, 0, stream>>>(ws+o_ss, head_w, head_b, out);
}
